// KNRM_28621662061083
// MI455X (gfx1250) — matrix operand order repaired and checked by reading, not compiled, not run on hardware
//
#include <hip/hip_runtime.h>
#include <hip/hip_bf16.h>
#include <math.h>

typedef __attribute__((ext_vector_type(16))) _Float16 v16h;
typedef __attribute__((ext_vector_type(8)))  float    v8f;

#define KNUM   21
#define BATCH  512
#define LQ     30
#define RD     200
#define DIMS   128
#define LQ_PAD 32
#define RD_PAD 208
#define NTILES 13   // ceil(200/16)
#define MTILES 2    // ceil(30/16)

// DPP16 row_shr butterfly: after 4 steps, lane 15 (and 31) holds the sum of its 16-lane row.
__device__ __forceinline__ float dpp_row_sum16(float x) {
    int t;
    t = __builtin_amdgcn_update_dpp(0, __float_as_int(x), 0x111, 0xf, 0xf, true); // row_shr:1
    x += __int_as_float(t);
    t = __builtin_amdgcn_update_dpp(0, __float_as_int(x), 0x112, 0xf, 0xf, true); // row_shr:2
    x += __int_as_float(t);
    t = __builtin_amdgcn_update_dpp(0, __float_as_int(x), 0x114, 0xf, 0xf, true); // row_shr:4
    x += __int_as_float(t);
    t = __builtin_amdgcn_update_dpp(0, __float_as_int(x), 0x118, 0xf, 0xf, true); // row_shr:8
    x += __int_as_float(t);
    return x;
}

__global__ __launch_bounds__(256) void knrm_predict_kernel(
    const float* __restrict__ emb,
    const float* __restrict__ W,
    const float* __restrict__ bias,
    const int*   __restrict__ q1, const int* __restrict__ d1,
    const int*   __restrict__ q2, const int* __restrict__ d2,
    float*       __restrict__ logits)
{
    // mu_i = (2i-19)/20 for i<20, clamped to 1.0 at i=20
    constexpr float MU[KNUM] = {
        -0.95f,-0.85f,-0.75f,-0.65f,-0.55f,-0.45f,-0.35f,-0.25f,-0.15f,-0.05f,
         0.05f, 0.15f, 0.25f, 0.35f, 0.45f, 0.55f, 0.65f, 0.75f, 0.85f, 0.95f, 1.0f };
    // s = sqrt((0.5/sigma^2)*log2(e)); exp term = 2^(-(s*(m-mu))^2), v_exp_f32 is 2^x
    constexpr float SQ  = 8.4932180571f;      // sigma = 0.1
    constexpr float SQX = 849.32180571f;      // sigma = 0.001 (exact-match kernel)

    __shared__ __align__(16) _Float16 sQ[LQ_PAD][DIMS];  //  8 KB
    __shared__ __align__(16) _Float16 sD[RD_PAD][DIMS];  // 52 KB
    __shared__ float sS[LQ_PAD][KNUM];                   // ~2.6 KB

    const int b    = blockIdx.x;
    const int pair = blockIdx.y;
    const int tid  = threadIdx.x;
    const int wave = tid >> 5;
    const int lane = tid & 31;

    const int* __restrict__ q = pair ? q2 : q1;
    const int* __restrict__ d = pair ? d2 : d1;

    // zero kernel accumulators
    for (int i = tid; i < LQ_PAD * KNUM; i += 256)
        (&sS[0][0])[i] = 0.f;

    // ---- gather + L2-normalize query rows -> f16 LDS (one wave per row) ----
    for (int row = wave; row < LQ_PAD; row += 8) {
        float x0 = 0.f, x1 = 0.f, x2 = 0.f, x3 = 0.f;
        if (row < LQ) {
            const int tok = q[b * LQ + row];
            const float4 v = *(const float4*)(emb + (size_t)tok * DIMS + lane * 4);
            x0 = v.x; x1 = v.y; x2 = v.z; x3 = v.w;
        }
        float ss = x0*x0 + x1*x1 + x2*x2 + x3*x3;
        #pragma unroll
        for (int off = 16; off >= 1; off >>= 1) ss += __shfl_xor(ss, off, 32);
        const float scale = 1.f / fmaxf(sqrtf(ss), 1e-12f);
        sQ[row][lane*4 + 0] = (_Float16)(x0 * scale);
        sQ[row][lane*4 + 1] = (_Float16)(x1 * scale);
        sQ[row][lane*4 + 2] = (_Float16)(x2 * scale);
        sQ[row][lane*4 + 3] = (_Float16)(x3 * scale);
    }
    // ---- gather + L2-normalize doc rows ----
    for (int row = wave; row < RD_PAD; row += 8) {
        float x0 = 0.f, x1 = 0.f, x2 = 0.f, x3 = 0.f;
        if (row < RD) {
            const int tok = d[b * RD + row];
            const float4 v = *(const float4*)(emb + (size_t)tok * DIMS + lane * 4);
            x0 = v.x; x1 = v.y; x2 = v.z; x3 = v.w;
        }
        float ss = x0*x0 + x1*x1 + x2*x2 + x3*x3;
        #pragma unroll
        for (int off = 16; off >= 1; off >>= 1) ss += __shfl_xor(ss, off, 32);
        const float scale = 1.f / fmaxf(sqrtf(ss), 1e-12f);
        sD[row][lane*4 + 0] = (_Float16)(x0 * scale);
        sD[row][lane*4 + 1] = (_Float16)(x1 * scale);
        sD[row][lane*4 + 2] = (_Float16)(x2 * scale);
        sD[row][lane*4 + 3] = (_Float16)(x3 * scale);
    }
    __syncthreads();

    // ---- tiled WMMA: M = sQ * sD^T, fused Gaussian kernel bank ----
    // A fragment (16x32 f16, ISA 7.12.2): lane m = lane&15; lanes>=16 K-base 8;
    //   vgpr g in 0..3 -> K pair base 2g; g in 4..7 -> 16+2(g-4)
    // B fragment (32x16 f16): lane n = lane&15; same K order as A (lanes>=16 K-base 8)
    const int am  = lane & 15;
    const int akb = (lane < 16) ? 0 : 8;
    const int bn  = lane & 15;
    const int bkb = (lane < 16) ? 0 : 8;

    for (int t = wave; t < MTILES * NTILES; t += 8) {
        const int mt = t / NTILES;
        const int nt = t % NTILES;
        const unsigned* __restrict__ aRow = (const unsigned*)&sQ[mt*16 + am][0];
        const unsigned* __restrict__ bRow = (const unsigned*)&sD[nt*16 + bn][0];

        union { v8f v; float f[8]; } C;
        C.v = (v8f){0.f,0.f,0.f,0.f,0.f,0.f,0.f,0.f};

        #pragma unroll
        for (int ks = 0; ks < 4; ++ks) {
            const int k0 = ks * 32;
            union { v16h v; unsigned u[8]; } A, Bf;
            #pragma unroll
            for (int g = 0; g < 8; ++g) {
                const int kp = akb + ((g < 4) ? (2*g) : (16 + 2*(g-4)));
                A.u[g]  = aRow[(k0 + kp) >> 1];
                Bf.u[g] = bRow[(k0 + bkb + ((g < 4) ? (2*g) : (16 + 2*(g-4)))) >> 1];
            }
            C.v = __builtin_amdgcn_wmma_f32_16x16x32_f16(
                false, A.v, false, Bf.v, (short)0, C.v, false, false);
        }

        // C layout: vgpr j -> M = j + 8*(lane>=16); N = lane&15
        const int  rg     = nt*16 + (lane & 15);
        const bool rvalid = (rg < RD);
        const int  mhi    = (lane >> 4) << 3;
        const bool writer = ((lane & 15) == 15);
        #pragma unroll
        for (int j = 0; j < 8; ++j) {
            const int  lg    = mt*16 + j + mhi;            // < 32, sS row always in-bounds
            const bool valid = rvalid && (lg < LQ);
            // poisoned m: every kernel's exp2 argument is hugely negative -> raw v_exp_f32
            // flushes to exact 0, so no per-kernel mask multiply and EXEC stays uniform.
            const float m = valid ? C.f[j] : 1024.f;
            float e21[KNUM];
            #pragma unroll
            for (int k = 0; k < KNUM; ++k) {
                const float s = (k == KNUM - 1) ? SQX : SQ;
                const float t2 = fmaf(m, s, -MU[k] * s);   // constant-folded -mu*s literal
                const float e  = __builtin_amdgcn_exp2f(-(t2 * t2)); // raw v_exp_f32, no fixup
                e21[k] = dpp_row_sum16(e);                  // VALU-only 16-lane reduction
            }
            if (writer) {                                   // lanes 15 & 31: one region, 21-op ds clause
                float* __restrict__ srow = &sS[lg][0];
                #pragma unroll
                for (int k = 0; k < KNUM; ++k)
                    atomicAdd(srow + k, e21[k]);            // ds_add_f32, immediate offsets
            }
        }
    }
    __syncthreads();

    // ---- km = sum_l log1p(S[l,k]); logit = km . W + b ----
    if (wave == 0) {
        float v = 0.f;
        if (lane < KNUM) {
            float km = 0.f;
            for (int l = 0; l < LQ; ++l) km += log1pf(sS[l][lane]);
            v = km * W[lane];
        }
        #pragma unroll
        for (int off = 16; off >= 1; off >>= 1) v += __shfl_xor(v, off, 32);
        if (lane == 0) logits[pair * BATCH + b] = v + bias[0];
    }
}

__global__ void knrm_sigmoid_kernel(const float* __restrict__ logits,
                                    float* __restrict__ out)
{
    const int i = blockIdx.x * blockDim.x + threadIdx.x;
    if (i < BATCH) {
        const float z = logits[i] - logits[BATCH + i];
        out[i] = 1.f / (1.f + __expf(-z));
    }
}

extern "C" void kernel_launch(void* const* d_in, const int* in_sizes, int n_in,
                              void* d_out, int out_size, void* d_ws, size_t ws_size,
                              hipStream_t stream) {
    (void)in_sizes; (void)n_in; (void)out_size; (void)ws_size;
    const float* emb  = (const float*)d_in[0];
    const float* W    = (const float*)d_in[1];
    const float* bias = (const float*)d_in[2];
    const int*   q1   = (const int*)d_in[3];
    const int*   dd1  = (const int*)d_in[4];
    const int*   q2   = (const int*)d_in[5];
    const int*   dd2  = (const int*)d_in[6];
    float* logits = (float*)d_ws;           // 2 * BATCH floats
    float* out    = (float*)d_out;          // BATCH floats

    dim3 grid(BATCH, 2);
    knrm_predict_kernel<<<grid, 256, 0, stream>>>(emb, W, bias, q1, dd1, q2, dd2, logits);
    knrm_sigmoid_kernel<<<(BATCH + 255) / 256, 256, 0, stream>>>(logits, out);
}
